// Net_35055523070561
// MI455X (gfx1250) — compile-verified, matrix operand order checked by reading, not run on hardware
//
#include <hip/hip_runtime.h>
#include <hip/hip_bf16.h>

typedef unsigned short u16;
typedef unsigned int   u32;

typedef __attribute__((ext_vector_type(16))) __bf16 v16bf;
typedef __attribute__((ext_vector_type(8)))  float  v8f;
typedef __attribute__((ext_vector_type(4)))  unsigned int u32x4;
typedef __attribute__((ext_vector_type(8)))  int i32x8;
typedef __attribute__((ext_vector_type(4)))  int i32x4;

#define BB 64
#define NN 128
#define KK 16
#define EE (BB * NN * KK)   // 131072 edges

__device__ __forceinline__ u16 f2bf(float f) {
    union { float f; u32 u; } v; v.f = f;
    u32 u = v.u;
    u32 r = (u + 0x7FFFu + ((u >> 16) & 1u)) >> 16;   // round-to-nearest-even
    return (u16)r;
}
__device__ __forceinline__ float bf2f(u16 h) {
    union { u32 u; float f; } v; v.u = ((u32)h) << 16;
    return v.f;
}

// ---------------------------------------------------------------------------
// TDM: async 2-D tensor load Global -> LDS.  D# per CDNA5 ISA §8.3/8.4:
//  group0: count=1, lds_addr, global_addr[56:0], type=2
//  group1: data_size=2B, tensor_dim0=dim0, tensor_dim1=rows,
//          tile_dim0=dim0, tile_dim1=rows, tensor_dim0_stride=dim0
// ---------------------------------------------------------------------------
__device__ __forceinline__ void tdm_load_2d(const void* gptr, unsigned lds_off,
                                            int dim0, int rows) {
    unsigned long long ga = (unsigned long long)gptr;
    u32x4 g0;
    g0.x = 1u;                                        // count=1 (valid descriptor)
    g0.y = lds_off;                                   // lds_addr
    g0.z = (unsigned)(ga & 0xFFFFFFFFu);              // global_addr[31:0]
    g0.w = (unsigned)((ga >> 32) & 0x01FFFFFFu) | (2u << 30);  // addr[56:32] | type=2
    unsigned w0 = 1u << 16;                           // data_size=1 (2 bytes)
    unsigned w1 = ((unsigned)dim0 & 0xFFFFu) << 16;               // tensor_dim0[15:0]
    unsigned w2 = (((unsigned)dim0 >> 16) & 0xFFFFu)
                | (((unsigned)rows & 0xFFFFu) << 16);             // dim0 hi | tensor_dim1 lo
    unsigned w3 = (((unsigned)rows >> 16) & 0xFFFFu)
                | (((unsigned)dim0 & 0xFFFFu) << 16);             // dim1 hi | tile_dim0
    unsigned w4 = (unsigned)rows & 0xFFFFu;                       // tile_dim1 (tile_dim2=0)
    unsigned w5 = (unsigned)dim0;                                 // tensor_dim0_stride[31:0]
    i32x8 g1 = { (int)w0, (int)w1, (int)w2, (int)w3,
                 (int)w4, (int)w5, 0, 0 };
    i32x4 gz = { 0, 0, 0, 0 };
#if defined(__clang_major__) && (__clang_major__ >= 23)
    i32x8 gz8 = { 0, 0, 0, 0, 0, 0, 0, 0 };
    __builtin_amdgcn_tensor_load_to_lds(g0, g1, gz, gz, gz8, 0);
#else
    __builtin_amdgcn_tensor_load_to_lds(g0, g1, gz, gz, 0);
#endif
}

// ---------------------------------------------------------------------------
// kNN: one block per batch event, one thread per point. x cached in LDS.
// ---------------------------------------------------------------------------
__global__ void knn_kernel(const float* __restrict__ x, int C, int* __restrict__ idx) {
    extern __shared__ float sx[];                // [NN * C]
    const int b = blockIdx.x, n = threadIdx.x;   // blockDim.x == NN
    const float* xb = x + (size_t)b * NN * C;
    for (int t = n; t < NN * C; t += NN) sx[t] = xb[t];
    __syncthreads();

    float bd[KK]; int bi[KK];
#pragma unroll
    for (int k = 0; k < KK; ++k) { bd[k] = 3.0e38f; bi[k] = 0; }

    for (int m = 0; m < NN; ++m) {
        if (m == n) continue;
        float d = 0.f;
        for (int c = 0; c < C; ++c) {
            float df = sx[n * C + c] - sx[m * C + c];
            d = fmaf(df, df, d);
        }
        if (d < bd[KK - 1]) {
            int p = KK - 1;
            while (p > 0 && bd[p - 1] > d) { bd[p] = bd[p - 1]; bi[p] = bi[p - 1]; --p; }
            bd[p] = d; bi[p] = m;
        }
    }
#pragma unroll
    for (int k = 0; k < KK; ++k) idx[((size_t)b * NN + n) * KK + k] = bi[k];
}

// ---------------------------------------------------------------------------
// Build edge messages [E, lda] bf16 : msg = [x_i , x_j - x_i], zero K-padding.
// ---------------------------------------------------------------------------
__global__ void build_msg(const float* __restrict__ x, const int* __restrict__ idx,
                          u16* __restrict__ msg, int C, int lda, long total) {
    long t = (long)blockIdx.x * blockDim.x + threadIdx.x;
    const long stride = (long)gridDim.x * blockDim.x;
    const int C2 = 2 * C;
    for (; t < total; t += stride) {
        long e = t / lda; int c = (int)(t % lda);
        float v;
        if (c >= C2) {
            v = 0.f;
        } else {
            long bn = e >> 4;                       // e = bn*16 + k
            long base_i = bn * C;
            if (c < C) {
                v = x[base_i + c];
            } else {
                int j = idx[e];
                long b = bn >> 7;                   // /NN
                long base_j = (b * NN + j) * (long)C;
                v = x[base_j + (c - C)] - x[base_i + (c - C)];
            }
        }
        msg[t] = f2bf(v);
    }
}

// ---------------------------------------------------------------------------
// WMMA GEMM: Y[M,Nc] = A[M,lda](bf16) x Wt[Nc,lda](bf16)^T + bias.
// Block = 4 waves x 128 threads; block owns a 64-row x 64-col output macrotile:
// wave w -> M-tile (blk*4+w); all waves share one 64-column N-group whose
// weights are TDM-loaded into LDS once.  Per K step: one A fragment (global),
// four B fragments (LDS, loaded into distinct registers so the 8 ds_load_b128
// issue as one clause), then 4 back-to-back WMMAs into 4 accumulators.
// Fused bias + per-channel sum/sumsq atomics for BatchNorm.
// ---------------------------------------------------------------------------
__global__ __launch_bounds__(128)
void wmma_gemm_bf16(const u16* __restrict__ A, const u16* __restrict__ Wt,
                    const float* __restrict__ bias, u16* __restrict__ Y,
                    float* __restrict__ sSum, float* __restrict__ sSqs,
                    int M, int Nc, int Kc, int lda, int doStats) {
    extern __shared__ u16 sw[];                  // [64 rows][lda] of Wt
    const int ngroups = Nc >> 6;
    const int ng = blockIdx.x % ngroups;
    const int mb = blockIdx.x / ngroups;
    const int waveId = threadIdx.x >> 5;
    const int lane = threadIdx.x & 31;

    if (waveId == 0) {
        tdm_load_2d(Wt + (size_t)(ng * 64) * lda, (unsigned)(size_t)sw, lda, 64);
        __builtin_amdgcn_s_wait_tensorcnt(0);
    }
    __syncthreads();

    const int rl = lane & 15, hi = lane >> 4;
    const int tm = mb * 4 + waveId;
    // 16-bit operand lane layout: lane<16 -> row rl, K {0..7,16..23};
    // lanes 16..31 -> row rl, K {8..15,24..31}.
    const u16* pA = A + (size_t)(tm * 16 + rl) * lda + (hi << 3);
    const u16* pB = sw + (size_t)rl * lda + (hi << 3);
    const int bstep = 16 * lda;                  // LDS stride between N-subtiles

    union F { uint4 q[2]; v16bf v; };
    v8f acc[4] = {v8f{}, v8f{}, v8f{}, v8f{}};
    for (int kk = 0; kk < Kc; kk += 32) {
        F a, b[4];
        a.q[0] = *(const uint4*)(pA + kk);
        a.q[1] = *(const uint4*)(pA + kk + 16);
        __builtin_prefetch(pA + kk + 64, 0, 1);  // global_prefetch_b8
#pragma unroll
        for (int t = 0; t < 4; ++t) {            // 8 ds_load_b128, one clause
            b[t].q[0] = *(const uint4*)(pB + t * bstep + kk);
            b[t].q[1] = *(const uint4*)(pB + t * bstep + kk + 16);
        }
#pragma unroll
        for (int t = 0; t < 4; ++t) {            // 4 back-to-back WMMAs
            acc[t] = __builtin_amdgcn_wmma_f32_16x16x32_bf16(
                         false, a.v, false, b[t].v, (short)0, acc[t], false, false);
        }
    }

    // D layout: VGPR v, lane -> row tm*16 + v + hi*8, col = subtile base + rl
#pragma unroll
    for (int t = 0; t < 4; ++t) {
        const int col = (ng * 4 + t) * 16 + rl;
        const float bval = bias[col];
        u16* py = Y + (size_t)(tm * 16 + (hi << 3)) * Nc + col;
        float s = 0.f, q = 0.f;
#pragma unroll
        for (int v = 0; v < 8; ++v) {
            float val = acc[t][v] + bval;
            py[(size_t)v * Nc] = f2bf(val);
            s += val; q = fmaf(val, val, q);
        }
        if (doStats) {
            s += __shfl_xor(s, 16, 32);          // combine the two half-tiles
            q += __shfl_xor(q, 16, 32);
            if (hi == 0) {
                atomicAdd(&sSum[col], s);
                atomicAdd(&sSqs[col], q);
            }
        }
    }
}

// ---------------------------------------------------------------------------
// In-place BatchNorm(batch stats) + ReLU on bf16 activations.
// ---------------------------------------------------------------------------
__global__ void bn_relu_apply(u16* __restrict__ Y,
                              const float* __restrict__ sSum, const float* __restrict__ sSqs,
                              const float* __restrict__ gamma, const float* __restrict__ beta,
                              long total, int Nc, float invE) {
    long t = (long)blockIdx.x * blockDim.x + threadIdx.x;
    const long stride = (long)gridDim.x * blockDim.x;
    for (; t < total; t += stride) {
        int c = (int)(t % Nc);
        float mu  = sSum[c] * invE;
        float var = fmaf(sSqs[c], invE, -mu * mu);
        float sc  = gamma[c] * rsqrtf(var + 1e-5f);
        float sh  = fmaf(-mu, sc, beta[c]);
        float v   = fmaf(bf2f(Y[t]), sc, sh);
        Y[t] = f2bf(v > 0.f ? v : 0.f);
    }
}

// ---------------------------------------------------------------------------
// Max over K neighbours -> node features (fp32 for next kNN).
// ---------------------------------------------------------------------------
__global__ void max_aggr(const u16* __restrict__ Y, float* __restrict__ xo,
                         long total, int C) {
    long t = (long)blockIdx.x * blockDim.x + threadIdx.x;
    const long stride = (long)gridDim.x * blockDim.x;
    for (; t < total; t += stride) {
        long bn = t / C; int c = (int)(t % C);
        const u16* p = Y + ((size_t)bn * KK) * C + c;
        float m = -3.0e38f;
#pragma unroll
        for (int k = 0; k < KK; ++k) m = fmaxf(m, bf2f(p[(size_t)k * C]));
        xo[t] = m;
    }
}

// Global max pool over N points -> pooled[B,256]
__global__ void global_pool(const float* __restrict__ x, float* __restrict__ pooled) {
    int t = blockIdx.x * blockDim.x + threadIdx.x;
    if (t >= BB * 256) return;
    int b = t >> 8, c = t & 255;
    float m = -3.0e38f;
    for (int n = 0; n < NN; ++n) m = fmaxf(m, x[((size_t)b * NN + n) * 256 + c]);
    pooled[t] = m;
}

__global__ void copy_points(const float* __restrict__ p, float* __restrict__ x) {
    int t = blockIdx.x * blockDim.x + threadIdx.x;
    if (t < BB * NN * 2) x[t] = p[t];
}

// Transpose + pad weights to bf16 [cout, lda]
__global__ void prep_wt(const float* __restrict__ W, u16* __restrict__ Wt,
                        int cin, int cout, int lda) {
    int t = blockIdx.x * blockDim.x + threadIdx.x;
    if (t >= cout * lda) return;
    int co = t / lda, kk = t % lda;
    float v = (kk < cin) ? W[(size_t)kk * cout + co] : 0.f;
    Wt[t] = f2bf(v);
}

// ---------------------------------------------------------------------------
// Tail: global-feature MLP (BN over 64 rows) + head MLP. One block, 64 threads,
// thread c owns output column c (BN stats reduce inside a thread).
// ---------------------------------------------------------------------------
__global__ __launch_bounds__(64)
void tail_kernel(const float* __restrict__ gf, const float* __restrict__ pooled,
                 const float* gW0, const float* gb0, const float* gG0, const float* gB0,
                 const float* gW1, const float* gb1,
                 const float* hW0, const float* hb0,
                 const float* hW1, const float* hb1,
                 const float* hW2, const float* hb2,
                 float* __restrict__ out) {
    __shared__ float z0[BB][65];
    __shared__ float z1[BB][65];
    const int c = threadIdx.x;                    // 0..63
    float col[BB];

    // gm layer0: Linear + BN(batch) + ReLU
    float mu = 0.f;
    for (int b = 0; b < BB; ++b) {
        float a = gb0[c];
#pragma unroll
        for (int ci = 0; ci < 4; ++ci) a = fmaf(gf[b * 4 + ci], gW0[ci * 64 + c], a);
        col[b] = a; mu += a;
    }
    mu *= (1.f / BB);
    float var = 0.f;
    for (int b = 0; b < BB; ++b) { float d = col[b] - mu; var = fmaf(d, d, var); }
    var *= (1.f / BB);
    float sc = gG0[c] * rsqrtf(var + 1e-5f), sh = fmaf(-mu, sc, gB0[c]);
    for (int b = 0; b < BB; ++b) { float v = fmaf(col[b], sc, sh); z0[b][c] = v > 0.f ? v : 0.f; }
    __syncthreads();

    // gm layer1: plain Linear -> z1
    for (int b = 0; b < BB; ++b) {
        float a = gb1[c];
        for (int ci = 0; ci < 64; ++ci) a = fmaf(z0[b][ci], gW1[ci * 64 + c], a);
        z1[b][c] = a;
    }
    __syncthreads();

    // head layer0: in = [pooled(256) | z1(64)] -> ReLU -> z0
    for (int b = 0; b < BB; ++b) {
        float a = hb0[c];
        for (int ci = 0; ci < 256; ++ci) a = fmaf(pooled[b * 256 + ci], hW0[ci * 64 + c], a);
        for (int ci = 0; ci < 64;  ++ci) a = fmaf(z1[b][ci], hW0[(256 + ci) * 64 + c], a);
        z0[b][c] = a > 0.f ? a : 0.f;
    }
    __syncthreads();

    // head layer1 -> ReLU -> z1
    for (int b = 0; b < BB; ++b) {
        float a = hb1[c];
        for (int ci = 0; ci < 64; ++ci) a = fmaf(z0[b][ci], hW1[ci * 64 + c], a);
        z1[b][c] = a > 0.f ? a : 0.f;
    }
    __syncthreads();

    // head layer2: [64,1]; thread b writes its row
    {
        int b = c;
        float a = hb2[0];
        for (int ci = 0; ci < 64; ++ci) a = fmaf(z1[b][ci], hW2[ci], a);
        out[b] = a;
    }
}

// ---------------------------------------------------------------------------
static inline int gsz(long total, int block) {
    long g = (total + block - 1) / block;
    if (g > 32768) g = 32768;
    if (g < 1) g = 1;
    return (int)g;
}

extern "C" void kernel_launch(void* const* d_in, const int* in_sizes, int n_in,
                              void* d_out, int out_size, void* d_ws, size_t ws_size,
                              hipStream_t stream) {
    (void)in_sizes; (void)n_in; (void)out_size; (void)ws_size;
    const float* points = (const float*)d_in[0];
    const float* gfeat  = (const float*)d_in[3];

    // ec_params: blocks at d_in[4..13], [14..23], [24..33]
    // each block: W0,b0,g0,be0, W1,b1,g1,be1, W2,b2
    const int ecBase[3] = {4, 14, 24};
    const int ecCin[3]  = {2, 64, 128};
    const int ecCout[3] = {64, 128, 256};
    const int ecLda0[3] = {32, 128, 256};   // 2*cin rounded up to 32

    // workspace carve-out
    char* ws = (char*)d_ws;
    size_t cur = 0;
    auto alloc = [&](size_t bytes) -> void* {
        size_t a = (cur + 255) & ~(size_t)255;
        cur = a + bytes;
        return (void*)(ws + a);
    };
    int*   idx    = (int*)  alloc((size_t)EE * 4);
    float* xbuf   = (float*)alloc((size_t)BB * NN * 256 * 4);
    u16*   bufA   = (u16*)  alloc((size_t)EE * 256 * 2);
    u16*   bufB   = (u16*)  alloc((size_t)EE * 256 * 2);
    float* stats  = (float*)alloc(512 * 4);           // sum[256] + sqs[256]
    float* pooled = (float*)alloc((size_t)BB * 256 * 4);
    u16*   wt[9];
    {
        int wi = 0;
        for (int blk = 0; blk < 3; ++blk) {
            int co = ecCout[blk];
            wt[wi++] = (u16*)alloc((size_t)co * ecLda0[blk] * 2);
            wt[wi++] = (u16*)alloc((size_t)co * co * 2);
            wt[wi++] = (u16*)alloc((size_t)co * co * 2);
        }
    }
    float* sSum = stats, *sSqs = stats + 256;

    // Prepare transposed/padded bf16 weights (deterministic, every call)
    {
        int wi = 0;
        for (int blk = 0; blk < 3; ++blk) {
            int base = ecBase[blk], co = ecCout[blk];
            int cins[3] = {2 * ecCin[blk], co, co};
            int ldas[3] = {ecLda0[blk], co, co};
            int woff[3] = {0, 4, 8};
            for (int l = 0; l < 3; ++l) {
                const float* W = (const float*)d_in[base + woff[l]];
                int tot = co * ldas[l];
                prep_wt<<<gsz(tot, 256), 256, 0, stream>>>(W, wt[wi], cins[l], co, ldas[l]);
                ++wi;
            }
        }
    }

    copy_points<<<gsz(BB * NN * 2, 256), 256, 0, stream>>>(points, xbuf);

    const float invE = 1.0f / (float)EE;
    for (int blk = 0; blk < 3; ++blk) {
        const int base = ecBase[blk];
        const int C = ecCin[blk], Co = ecCout[blk], lda0 = ecLda0[blk];
        const float* b0  = (const float*)d_in[base + 1];
        const float* g0  = (const float*)d_in[base + 2];
        const float* be0 = (const float*)d_in[base + 3];
        const float* b1  = (const float*)d_in[base + 5];
        const float* g1  = (const float*)d_in[base + 6];
        const float* be1 = (const float*)d_in[base + 7];
        const float* b2  = (const float*)d_in[base + 9];
        u16 *w0 = wt[blk * 3], *w1 = wt[blk * 3 + 1], *w2 = wt[blk * 3 + 2];

        // 1) kNN graph on current features
        knn_kernel<<<BB, NN, (size_t)NN * C * 4, stream>>>(xbuf, C, idx);

        // 2) edge messages -> bufA (bf16, K-padded)
        build_msg<<<gsz((long)EE * lda0, 256), 256, 0, stream>>>(
            xbuf, idx, bufA, C, lda0, (long)EE * lda0);

        const int blocks = (EE / 64) * (Co / 64);

        // 3) layer0: GEMM (TDM weights -> LDS) + stats, BN+ReLU in place
        hipMemsetAsync(stats, 0, 512 * 4, stream);
        wmma_gemm_bf16<<<blocks, 128, (size_t)64 * lda0 * 2, stream>>>(
            bufA, w0, b0, bufB, sSum, sSqs, EE, Co, lda0, lda0, 1);
        bn_relu_apply<<<gsz((long)EE * Co, 256), 256, 0, stream>>>(
            bufB, sSum, sSqs, g0, be0, (long)EE * Co, Co, invE);

        // 4) layer1
        hipMemsetAsync(stats, 0, 512 * 4, stream);
        wmma_gemm_bf16<<<blocks, 128, (size_t)64 * Co * 2, stream>>>(
            bufB, w1, b1, bufA, sSum, sSqs, EE, Co, Co, Co, 1);
        bn_relu_apply<<<gsz((long)EE * Co, 256), 256, 0, stream>>>(
            bufA, sSum, sSqs, g1, be1, (long)EE * Co, Co, invE);

        // 5) layer2 (plain linear)
        wmma_gemm_bf16<<<blocks, 128, (size_t)64 * Co * 2, stream>>>(
            bufA, w2, b2, bufB, sSum, sSqs, EE, Co, Co, Co, 0);

        // 6) max over K -> node features
        max_aggr<<<gsz((long)BB * NN * Co, 256), 256, 0, stream>>>(
            bufB, xbuf, (long)BB * NN * Co, Co);
    }

    // global max pool over points
    global_pool<<<gsz(BB * 256, 256), 256, 0, stream>>>(xbuf, pooled);

    // tail MLPs -> out[64]
    tail_kernel<<<1, 64, 0, stream>>>(
        gfeat, pooled,
        (const float*)d_in[34], (const float*)d_in[35],
        (const float*)d_in[36], (const float*)d_in[37],
        (const float*)d_in[38], (const float*)d_in[39],
        (const float*)d_in[40], (const float*)d_in[41],
        (const float*)d_in[42], (const float*)d_in[43],
        (const float*)d_in[44], (const float*)d_in[45],
        (float*)d_out);
}
